// VGAEencoder_4071628996675
// MI455X (gfx1250) — compile-verified, matrix operand order checked by reading, not run on hardware
//
#include <hip/hip_runtime.h>

// Problem constants (fixed by the reference harness)
#define GN   50000      // nodes  (== 3125 * 16, exact 16-row tiles)
#define GE   800000     // edges
#define IN_C 128
#define HID  128        // H = 2*OUT
#define OUTC 64

typedef __attribute__((ext_vector_type(2)))  float    v2f;
typedef __attribute__((ext_vector_type(8)))  float    v8f;
typedef __attribute__((ext_vector_type(16))) _Float16 v16h;

// ---------------------------------------------------------------------------
// Degree / normalization kernels
// ---------------------------------------------------------------------------
__global__ __launch_bounds__(256) void k_init_deg(float* deg) {
    int i = blockIdx.x * blockDim.x + threadIdx.x;
    if (i < GN) deg[i] = 1.0f;                       // self-loop
}

__global__ __launch_bounds__(256) void k_scatter_deg(const long long* __restrict__ ei,
                                                     float* deg) {
    int e = blockIdx.x * blockDim.x + threadIdx.x;
    if (e < GE) atomicAdd(&deg[(int)ei[GE + e]], 1.0f);   // in-degree on dst
}

__global__ __launch_bounds__(256) void k_dinv(const float* __restrict__ deg, float* dinv) {
    int i = blockIdx.x * blockDim.x + threadIdx.x;
    if (i < GN) dinv[i] = rsqrtf(deg[i]);
}

__global__ __launch_bounds__(256) void k_edge_norm(const long long* __restrict__ ei,
                                                   const float* __restrict__ dinv,
                                                   float* nrm) {
    int e = blockIdx.x * blockDim.x + threadIdx.x;
    if (e < GE) nrm[e] = dinv[(int)ei[e]] * dinv[(int)ei[GE + e]];
}

// ---------------------------------------------------------------------------
// WMMA GEMM:  C[M, NDIM] = A[M, KDIM] @ B[KDIM, NDIM]   (row-major, f32)
// One wave32 owns a 16x16 output tile. RELU fuses max(0,.) into A-loads.
// Preferred path: exact f32 V_WMMA_F32_16X16X4_F32 (compute is not the
// bottleneck here, so keep reference precision). Fallback: confirmed
// f16-input / f32-accumulate WMMA.
// ---------------------------------------------------------------------------
template <int KDIM, int NDIM, bool RELU>
__global__ __launch_bounds__(128) void k_gemm_wmma(const float* __restrict__ A,
                                                   const float* __restrict__ B,
                                                   float* __restrict__ C) {
    constexpr int NT = NDIM / 16;
    const int lane = threadIdx.x & 31;
    const int wave = threadIdx.x >> 5;
    const int gw   = blockIdx.x * 4 + wave;     // 4 waves / block, exact grids
    const int tm   = gw / NT;                   // 16-row tile
    const int tn   = gw % NT;                   // 16-col tile
    const int half = lane & 15;
    const int hi   = lane >> 4;                 // 0: lanes 0-15, 1: lanes 16-31

    const int arow = tm * 16 + half;
    const int bcol = tn * 16 + half;

    v8f acc = {};

#if __has_builtin(__builtin_amdgcn_wmma_f32_16x16x4_f32)
    // f32 A 16x4 layout: lanes 0-15 carry K=k0,k0+1; lanes 16-31 carry K=k0+2,k0+3
    const int koff = hi * 2;
#pragma unroll
    for (int k0 = 0; k0 < KDIM; k0 += 4) {
        float a0 = A[arow * KDIM + k0 + koff];
        float a1 = A[arow * KDIM + k0 + koff + 1];
        if (RELU) { a0 = fmaxf(a0, 0.0f); a1 = fmaxf(a1, 0.0f); }
        v2f a; a.x = a0; a.y = a1;
        v2f b; b.x = B[(k0 + koff) * NDIM + bcol];
               b.y = B[(k0 + koff + 1) * NDIM + bcol];
        acc = __builtin_amdgcn_wmma_f32_16x16x4_f32(
                  false, a, false, b, (short)0, acc, false, false);
    }
#else
    // f16 A 16x32 layout: lanes 0-15 hold K 0-7 & 16-23, lanes 16-31 hold 8-15 & 24-31
#pragma unroll
    for (int k0 = 0; k0 < KDIM; k0 += 32) {
        v16h a, b;
#pragma unroll
        for (int j = 0; j < 16; ++j) {
            int k = k0 + ((j < 8) ? j : j + 8) + hi * 8;
            float av = A[arow * KDIM + k];
            if (RELU) av = fmaxf(av, 0.0f);
            a[j] = (_Float16)av;
            b[j] = (_Float16)B[k * NDIM + bcol];
        }
        acc = __builtin_amdgcn_wmma_f32_16x16x32_f16(
                  false, a, false, b, (short)0, acc, false, false);
    }
#endif

    // D layout: VGPR r -> row (hi ? r+8 : r) of the tile, col = lane&15
#pragma unroll
    for (int r = 0; r < 8; ++r) {
        int m = tm * 16 + hi * 8 + r;
        C[m * NDIM + bcol] = acc[r];
    }
}

// ---------------------------------------------------------------------------
// agg[i,f] = h[i,f] * dinv[i]^2 + bias[f]      (self-loop term + bias)
// ---------------------------------------------------------------------------
template <int F>
__global__ __launch_bounds__(256) void k_self_bias(const float* __restrict__ h,
                                                   const float* __restrict__ dinv,
                                                   const float* __restrict__ bias,
                                                   float* __restrict__ agg) {
    int t = blockIdx.x * blockDim.x + threadIdx.x;
    if (t >= GN * F) return;
    int r = t / F, f = t & (F - 1);
    float di = dinv[r];
    agg[t] = h[t] * di * di + bias[f];
}

// ---------------------------------------------------------------------------
// agg[dst,f] += h[src,f] * norm[e]   (edge scatter; everything L2-resident)
// ---------------------------------------------------------------------------
template <int F>
__global__ __launch_bounds__(256) void k_scatter(const long long* __restrict__ ei,
                                                 const float* __restrict__ nrm,
                                                 const float* __restrict__ h,
                                                 float* agg) {
    int t = blockIdx.x * blockDim.x + threadIdx.x;   // GE*F <= 102.4M, fits int
    if (t >= GE * F) return;
    int e = t / F, f = t & (F - 1);
    int src = (int)ei[e];
    int dst = (int)ei[GE + e];
    atomicAdd(&agg[dst * F + f], h[src * F + f] * nrm[e]);
}

// ---------------------------------------------------------------------------
// logstd = mu (source bug: same weights twice), zeta = mu + eps*exp(logstd)
// ---------------------------------------------------------------------------
__global__ __launch_bounds__(256) void k_finalize(const float* __restrict__ mu,
                                                  const float* __restrict__ eps,
                                                  float* __restrict__ logstd,
                                                  float* __restrict__ zeta) {
    int i = blockIdx.x * blockDim.x + threadIdx.x;
    if (i >= GN * OUTC) return;
    float m = mu[i];
    logstd[i] = m;
    zeta[i]   = fmaf(eps[i], expf(m), m);
}

// ---------------------------------------------------------------------------
extern "C" void kernel_launch(void* const* d_in, const int* in_sizes, int n_in,
                              void* d_out, int out_size, void* d_ws, size_t ws_size,
                              hipStream_t stream) {
    const float*     x   = (const float*)d_in[0];
    const long long* ei  = (const long long*)d_in[1];   // int64 [2,E]
    const float*     W1  = (const float*)d_in[2];
    const float*     b1  = (const float*)d_in[3];
    const float*     Wmu = (const float*)d_in[4];
    const float*     bmu = (const float*)d_in[5];
    // d_in[6]/d_in[7] (Wls/bls) intentionally unused: reference bug reuses Wmu/bmu
    const float*     eps = (const float*)d_in[8];

    float* out    = (float*)d_out;
    float* mu     = out;                              // [N,64]
    float* logstd = out + (size_t)GN * OUTC;          // [N,64]
    float* zeta   = out + (size_t)2 * GN * OUTC;      // [N,64]

    float* ws   = (float*)d_ws;
    float* deg  = ws;                                 // [N]
    float* dinv = ws + GN;                            // [N]
    float* nrm  = ws + 2 * GN;                        // [E]
    float* h0   = ws + 2 * GN + GE;                   // [N,128]
    float* agg1 = h0 + (size_t)GN * HID;              // [N,128]
    float* m0   = h0;                                 // reuse h0 after layer 1

    const int T = 256;
    k_init_deg   <<<(GN + T - 1) / T, T, 0, stream>>>(deg);
    k_scatter_deg<<<(GE + T - 1) / T, T, 0, stream>>>(ei, deg);
    k_dinv       <<<(GN + T - 1) / T, T, 0, stream>>>(deg, dinv);
    k_edge_norm  <<<(GE + T - 1) / T, T, 0, stream>>>(ei, dinv, nrm);

    // Layer 1: h0 = X @ W1 ; agg1 = scatter + self + b1 ; (ReLU fused into GEMM2 loads)
    k_gemm_wmma<IN_C, HID, false>
        <<<(GN / 16) * (HID / 16) / 4, 128, 0, stream>>>(x, W1, h0);
    k_self_bias<HID><<<(GN * HID + T - 1) / T, T, 0, stream>>>(h0, dinv, b1, agg1);
    k_scatter<HID>  <<<(GE * HID + T - 1) / T, T, 0, stream>>>(ei, nrm, h0, agg1);

    // Layer 2: m0 = relu(agg1) @ Wmu ; mu = scatter + self + bmu  (into d_out)
    k_gemm_wmma<HID, OUTC, true>
        <<<(GN / 16) * (OUTC / 16) / 4, 128, 0, stream>>>(agg1, Wmu, m0);
    k_self_bias<OUTC><<<(GN * OUTC + T - 1) / T, T, 0, stream>>>(m0, dinv, bmu, mu);
    k_scatter<OUTC>  <<<(GE * OUTC + T - 1) / T, T, 0, stream>>>(ei, nrm, m0, mu);

    k_finalize<<<(GN * OUTC + T - 1) / T, T, 0, stream>>>(mu, eps, logstd, zeta);
}
